// VQVAE_60997125538173
// MI455X (gfx1250) — matrix operand order repaired and checked by reading, not compiled, not run on hardware
//
#include <hip/hip_runtime.h>
#include <stdint.h>

// ---------------------------------------------------------------------------
// VQ-VAE forward for gfx1250 (MI455X).
//  * NHWC bf16 feature maps, WMMA bf16 implicit-GEMM convs (f32 accumulate)
//  * Transposed convs decomposed into 4 parity sub-convs -> same WMMA kernel
//  * Per-tap weight tiles staged to LDS via async global->LDS b128 copies
//  * VQ nearest-code search via fp32 WMMA (v_wmma_f32_16x16x4_f32) + ds_min_u64
//  * Deterministic loss reduction (block partials + serial finalize)
// ---------------------------------------------------------------------------

typedef __bf16 bfx16 __attribute__((ext_vector_type(16)));
typedef float  fx8   __attribute__((ext_vector_type(8)));
typedef float  fx2   __attribute__((ext_vector_type(2)));
typedef int    v4i   __attribute__((vector_size(16)));

union BF16x16 { bfx16 v; uint4 q[2]; };

__device__ __forceinline__ unsigned short f2bf(float f) {
    unsigned u = __float_as_uint(f);
    unsigned r = (u + 0x7fffu + ((u >> 16) & 1u)) >> 16;   // round-nearest-even
    return (unsigned short)r;
}
__device__ __forceinline__ float bf2f(unsigned short h) {
    return __uint_as_float(((unsigned)h) << 16);
}

// 16-byte global -> LDS copy. Prefer the CDNA5 async-to-LDS data path
// (ASYNCcnt-tracked, bypasses VGPRs); fall back to a register copy.
__device__ __forceinline__ void g2lds16(const unsigned short* g, unsigned short* l) {
#if defined(__gfx1250__) && __has_builtin(__builtin_amdgcn_global_load_async_to_lds_b128)
    __builtin_amdgcn_global_load_async_to_lds_b128(
        (__attribute__((address_space(1))) v4i*)(unsigned short*)g,
        (__attribute__((address_space(3))) v4i*)l, 0, 0);
#else
    *(uint4*)l = *(const uint4*)g;
#endif
}
__device__ __forceinline__ void g2lds_wait() {
#if defined(__gfx1250__) && __has_builtin(__builtin_amdgcn_global_load_async_to_lds_b128)
#if __has_builtin(__builtin_amdgcn_s_wait_asynccnt)
    __builtin_amdgcn_s_wait_asynccnt(0);
#else
    asm volatile("s_wait_asynccnt 0" ::: "memory");
#endif
#endif
}

// ---------------------------------------------------------------------------
// Workspace layout (bytes)
// ---------------------------------------------------------------------------
static constexpr size_t OFF_C2   = 0;                       //  512 f32 codebook norms
static constexpr size_t OFF_VQP  = 4096;                    // 1024 f32 vq partials
static constexpr size_t OFF_RECP = 8192;                    // 12288 f32 recon partials
static constexpr size_t OFF_WE2  = 1ull << 16;              // 16*128*128 bf16
static constexpr size_t OFF_WE3  = OFF_WE2 + 524288ull;     // 9*64*128 bf16
static constexpr size_t OFF_WD1  = OFF_WE3 + 147456ull;     // 16*128*64 bf16
static constexpr size_t OFF_WD2  = OFF_WD1 + 262144ull;     // 16*128*128 bf16
static constexpr size_t OFF_H1   = 2ull << 20;              // 64x64x64x128 bf16
static constexpr size_t OFF_H2   = OFF_H1 + 67108864ull;    // 64x32x32x128 bf16
static constexpr size_t OFF_ZE   = OFF_H2 + 16777216ull;    // 64x32x32x64 f32
static constexpr size_t OFF_ZQ   = OFF_ZE + 16777216ull;    // 64x32x32x64 bf16
static constexpr size_t OFF_D1   = OFF_ZQ + 8388608ull;     // 64x64x64x128 bf16
static constexpr size_t OFF_D2   = OFF_D1 + 67108864ull;    // 64x128x128x128 bf16
// total ~447 MB

// ---------------------------------------------------------------------------
// Weight transform: OIHW / IOHW fp32 -> [tap][OC][C] bf16  (B-ready layout)
// ---------------------------------------------------------------------------
__global__ __launch_bounds__(256) void wt_transform_kernel(
    const float* __restrict__ w, unsigned short* __restrict__ out,
    int C, int OC, int Kk, int iohw)
{
    int idx = blockIdx.x * 256 + threadIdx.x;
    int total = Kk * Kk * C * OC;
    if (idx >= total) return;
    int c  = idx % C;
    int oc = (idx / C) % OC;
    int t  = idx / (C * OC);
    int ky = t / Kk, kx = t % Kk;
    float v = iohw ? w[((c * OC + oc) * Kk + ky) * Kk + kx]
                   : w[((oc * C + c) * Kk + ky) * Kk + kx];
    out[idx] = f2bf(v);
}

// ---------------------------------------------------------------------------
// Encoder conv1: 3->128, k4 s2 p1, relu.  x f32 NCHW -> h1 bf16 NHWC
// ---------------------------------------------------------------------------
__global__ __launch_bounds__(256) void enc1_kernel(
    const float* __restrict__ x, const float* __restrict__ w,
    const float* __restrict__ bias, unsigned short* __restrict__ h1)
{
    __shared__ float lw[128 * 3 * 16];
    int tid = threadIdx.x;
    for (int i = tid; i < 6144; i += 256) lw[i] = w[i];
    __syncthreads();

    int idx = (blockIdx.x << 8) + tid;          // [B=64][64][64][128]
    int oc = idx & 127;
    int xo = (idx >> 7) & 63;
    int yo = (idx >> 13) & 63;
    int b  = idx >> 19;
    float s = bias[oc];
    const float* xb = x + (((size_t)b * 3) << 14);
    const float* wp = lw + oc * 48;
    for (int ic = 0; ic < 3; ++ic)
        for (int ky = 0; ky < 4; ++ky) {
            int iy = (yo << 1) + ky - 1;
            if (iy < 0 || iy >= 128) continue;
            for (int kx = 0; kx < 4; ++kx) {
                int ix = (xo << 1) + kx - 1;
                if (ix < 0 || ix >= 128) continue;
                s += xb[(ic << 14) + (iy << 7) + ix] * wp[ic * 16 + ky * 4 + kx];
            }
        }
    h1[idx] = f2bf(fmaxf(s, 0.f));
}

// ---------------------------------------------------------------------------
// Generic implicit-GEMM conv via WMMA bf16.
//   in : NHWC bf16 [B, inH, inW, C]        (C = 64 or 128)
//   wt : [tap][OC][C] bf16                 (OC = 64 or 128)
//   Logical out grid ly x lx (powers of 2); output pixel:
//     Y = y*ostride + ooffy, X = x*ostride + ooffx  (NHWC, OC channels)
//   Tap table: 8 bits/tap: [1:0]=dy+1 [3:2]=dx+1 [7:4]=weight-tap index.
//   Input coord: iy = y*stride + dy (zero outside).
// One block = 8 waves x 16 pixels = 128 pixels, all OC channels.
// Per tap: whole [OC][C] weight tile async-staged to LDS once, then the
// channel loop runs barrier-free (2 global b128 A-loads + up to 16 ds b128 B-loads
// + up to 8 WMMAs per step).
// ---------------------------------------------------------------------------
__global__ __launch_bounds__(256) void conv_wmma_kernel(
    const unsigned short* __restrict__ in,
    const unsigned short* __restrict__ wt,
    const float* __restrict__ bias,
    unsigned short* __restrict__ out_bf, float* __restrict__ out_f32,
    int inH, int inW, int C,
    int lxBits, int bpiBits,
    int outH, int outW, int OC,
    int T, unsigned long long taps0, unsigned long long taps1,
    int stride, int ostride, int ooffy, int ooffx, int relu)
{
    __shared__ __align__(16) unsigned short lwt[16384];   // [oc][C] <= 32 KB

    int tid   = threadIdx.x;
    int wave  = tid >> 5;
    int lane  = tid & 31;
    int hl    = lane >> 4;                      // K-half selector
    int nlane = lane & 15;
    int lxMask = (1 << lxBits) - 1;
    int bpiMask = (1 << bpiBits) - 1;
    int img = blockIdx.x >> bpiBits;
    int p0  = ((blockIdx.x & bpiMask) << 7) + (wave << 4);

    // This lane's A-matrix row pixel
    int p  = p0 + nlane;
    int py = p >> lxBits, px = p & lxMask;

    const fx8 zero8 = {0, 0, 0, 0, 0, 0, 0, 0};
    fx8 acc[8];
#pragma unroll
    for (int i = 0; i < 8; ++i) acc[i] = zero8;

    size_t inImgBase = (size_t)img * inH * inW * C;
    int tileElems = OC * C;                     // bf16 elements per tap tile

    for (int t = 0; t < T; ++t) {
        unsigned tb = (unsigned)((t < 8 ? (taps0 >> (8 * t))
                                        : (taps1 >> (8 * (t - 8)))) & 0xffull);
        int dy  = (int)(tb & 3u) - 1;
        int dx  = (int)((tb >> 2) & 3u) - 1;
        int wti = (int)(tb >> 4);
        int iy = py * stride + dy;
        int ix = px * stride + dx;
        bool ok = (iy >= 0) & (iy < inH) & (ix >= 0) & (ix < inW);
        size_t rowBase = 0;
        if (ok) {
            rowBase = inImgBase + (size_t)(iy * inW + ix) * C;
            __builtin_prefetch((const void*)(in + rowBase), 0, 1);
        }

        // ---- stage this tap's full weight tile [OC][C] into LDS ----
        __syncthreads();                        // previous tile no longer in use
        {
            const unsigned short* wp = wt + (size_t)wti * tileElems;
            int nchunk = tileElems >> 3;        // 16-byte chunks
            for (int i = tid; i < nchunk; i += 256)
                g2lds16(wp + (i << 3), lwt + (i << 3));
        }
        g2lds_wait();
        __syncthreads();

        // ---- K loop over 32-channel chunks, barrier-free ----
        for (int c0 = 0; c0 < C; c0 += 32) {
            // A fragment: hl=0 -> K {0..7,16..23}; hl=1 -> K {8..15,24..31}
            BF16x16 aU;
            if (ok) {
                const uint4* ap = (const uint4*)(in + rowBase + c0 + 8 * hl);
                aU.q[0] = ap[0];                // K = 8*hl + 0..7
                aU.q[1] = ap[2];                // K = 16 + 8*hl + 0..7
            } else {
                aU.q[0] = make_uint4(0, 0, 0, 0);
                aU.q[1] = make_uint4(0, 0, 0, 0);
            }

#pragma unroll
            for (int nt = 0; nt < 8; ++nt) {
                if ((nt << 4) < OC) {
                    BF16x16 bU;
                    const uint4* bp = (const uint4*)(lwt +
                        (((nt << 4) + nlane) * C + c0 + (hl << 3)));
                    bU.q[0] = bp[0];
                    bU.q[1] = bp[2];
                    acc[nt] = __builtin_amdgcn_wmma_f32_16x16x32_bf16(
                        false, aU.v, false, bU.v, (short)0, acc[nt], false, false);
                }
            }
        }
    }

    // Epilogue: bias (+relu), store NHWC (bf16 or f32)
#pragma unroll
    for (int nt = 0; nt < 8; ++nt) {
        if ((nt << 4) < OC) {
            int oc = (nt << 4) + nlane;
            float bv = bias[oc];
#pragma unroll
            for (int r = 0; r < 8; ++r) {
                int pr = p0 + r + (hl << 3);
                int yy = pr >> lxBits, xx = pr & lxMask;
                int Y = yy * ostride + ooffy;
                int X = xx * ostride + ooffx;
                float v = acc[nt][r] + bv;
                if (relu) v = fmaxf(v, 0.f);
                size_t o = (((size_t)img * outH + Y) * outW + X) * OC + oc;
                if (out_bf) out_bf[o] = f2bf(v);
                else        out_f32[o] = v;
            }
        }
    }
}

// ---------------------------------------------------------------------------
// Codebook squared norms
// ---------------------------------------------------------------------------
__global__ __launch_bounds__(256) void cbnorm_kernel(
    const float* __restrict__ cb, float* __restrict__ c2)
{
    int i = blockIdx.x * 256 + threadIdx.x;
    if (i >= 512) return;
    const float* c = cb + (size_t)i * 64;
    float s = 0.f;
    for (int k = 0; k < 64; ++k) s += c[k] * c[k];
    c2[i] = s;
}

// ---------------------------------------------------------------------------
// Vector quantization: fp32 WMMA z.cT, argmin via sortable-key ds_min_u64,
// write z_q (bf16 NHWC) and per-block sum of (z_e - z_q)^2.
// Block = 128 threads (4 waves), 16 rows/wave, 64 rows/block.
// ---------------------------------------------------------------------------
__global__ __launch_bounds__(128) void vq_kernel(
    const float* __restrict__ ze, const float* __restrict__ cb,
    const float* __restrict__ c2, unsigned short* __restrict__ zq,
    float* __restrict__ vq_part)
{
    __shared__ unsigned long long best[64];
    __shared__ float red[128];
    int tid = threadIdx.x, wave = tid >> 5, lane = tid & 31;
    int hl = lane >> 4, nlane = lane & 15;
    int rowBlock = blockIdx.x << 6;
    int row0 = rowBlock + (wave << 4);

    if (lane < 16) best[(wave << 4) + lane] = ~0ull;
    __syncthreads();

    // A fragments: 16 K-chunks of 4. lane half picks K pair within chunk.
    fx2 a[16];
    const float* zr = ze + (size_t)(row0 + nlane) * 64;
#pragma unroll
    for (int kk = 0; kk < 16; ++kk) {
        a[kk].x = zr[kk * 4 + 2 * hl];
        a[kk].y = zr[kk * 4 + 2 * hl + 1];
    }

    unsigned long long bk[8];
#pragma unroll
    for (int r = 0; r < 8; ++r) bk[r] = ~0ull;

    for (int ct = 0; ct < 32; ++ct) {           // 32 tiles of 16 codes
        int c0 = ct << 4;
        const float* cr = cb + (size_t)(c0 + nlane) * 64;
        fx8 d = {0, 0, 0, 0, 0, 0, 0, 0};
#pragma unroll
        for (int kk = 0; kk < 16; ++kk) {
            fx2 b;
            b.x = cr[kk * 4 + 2 * hl];
            b.y = cr[kk * 4 + 2 * hl + 1];
            d = __builtin_amdgcn_wmma_f32_16x16x4_f32(
                false, a[kk], false, b, (short)0, d, false, false);
        }
        int n = c0 + nlane;
        float cn = c2[n];
#pragma unroll
        for (int r = 0; r < 8; ++r) {
            float val = cn - 2.f * d[r];        // ||z||^2 constant per row: drop
            unsigned u = __float_as_uint(val);
            u = (u & 0x80000000u) ? ~u : (u | 0x80000000u);
            unsigned long long key = ((unsigned long long)u << 32) | (unsigned)n;
            bk[r] = (key < bk[r]) ? key : bk[r];
        }
    }
#pragma unroll
    for (int r = 0; r < 8; ++r)
        atomicMin(&best[(wave << 4) + r + (hl << 3)], bk[r]);
    __syncthreads();

    float partial = 0.f;
    if (tid < 64) {
        int row = rowBlock + tid;
        int idx = (int)(best[tid] & 0xffffffffull);
        const float* z = ze + (size_t)row * 64;
        const float* c = cb + (size_t)idx * 64;
        unsigned short* q = zq + (size_t)row * 64;
        for (int k = 0; k < 64; ++k) {
            float cv = c[k];
            float df = z[k] - cv;
            partial += df * df;
            q[k] = f2bf(cv);
        }
    }
    red[tid] = partial;
    __syncthreads();
    for (int s = 64; s > 0; s >>= 1) {
        if (tid < s) red[tid] += red[tid + s];
        __syncthreads();
    }
    if (tid == 0) vq_part[blockIdx.x] = red[0];
}

// ---------------------------------------------------------------------------
// Final conv 128->3 k3 p1 (d2 NHWC bf16 -> x_recon NCHW f32 in d_out),
// fused recon-loss partial.
// ---------------------------------------------------------------------------
__global__ __launch_bounds__(256) void dec3_kernel(
    const unsigned short* __restrict__ d2, const float* __restrict__ w,
    const float* __restrict__ bias, const float* __restrict__ x,
    float* __restrict__ out, float* __restrict__ rec_part)
{
    __shared__ float lw[3 * 128 * 9];
    __shared__ float red[256];
    int tid = threadIdx.x;
    for (int i = tid; i < 3456; i += 256) lw[i] = w[i];
    __syncthreads();

    int idx = (blockIdx.x << 8) + tid;          // NCHW [64][3][128][128]
    int ox = idx & 127;
    int oy = (idx >> 7) & 127;
    int t  = idx >> 14;
    int oc = t % 3;
    int b  = t / 3;

    float s = bias[oc];
    for (int ky = 0; ky < 3; ++ky) {
        int iy = oy + ky - 1;
        if (iy < 0 || iy >= 128) continue;
        for (int kx = 0; kx < 3; ++kx) {
            int ix = ox + kx - 1;
            if (ix < 0 || ix >= 128) continue;
            const unsigned short* p = d2 + ((((size_t)b << 7) + iy) * 128 + ix) * 128;
            const float* wp = lw + oc * 1152 + ky * 3 + kx;
            for (int ic = 0; ic < 128; ++ic)
                s += bf2f(p[ic]) * wp[ic * 9];
        }
    }
    out[idx] = s;
    float df = s - x[idx];
    red[tid] = df * df;
    __syncthreads();
    for (int st = 128; st > 0; st >>= 1) {
        if (tid < st) red[tid] += red[tid + st];
        __syncthreads();
    }
    if (tid == 0) rec_part[blockIdx.x] = red[0];
}

// ---------------------------------------------------------------------------
// Deterministic loss finalize
// ---------------------------------------------------------------------------
__global__ __launch_bounds__(256) void finalize_kernel(
    const float* __restrict__ vqp, const float* __restrict__ recp,
    float* __restrict__ out3)
{
    __shared__ float red[256];
    int tid = threadIdx.x;
    float s = 0.f;
    for (int i = tid; i < 1024; i += 256) s += vqp[i];
    red[tid] = s;
    __syncthreads();
    for (int st = 128; st > 0; st >>= 1) {
        if (tid < st) red[tid] += red[tid + st];
        __syncthreads();
    }
    float vqsum = red[0];
    __syncthreads();
    s = 0.f;
    for (int i = tid; i < 12288; i += 256) s += recp[i];
    red[tid] = s;
    __syncthreads();
    for (int st = 128; st > 0; st >>= 1) {
        if (tid < st) red[tid] += red[tid + st];
        __syncthreads();
    }
    if (tid == 0) {
        float rec = red[0] / 3145728.f;
        float vq  = 1.25f * vqsum / 4194304.f;  // (1 + beta) * mse
        out3[0] = rec + vq;
        out3[1] = rec;
        out3[2] = vq;
    }
}

// ---------------------------------------------------------------------------
// Host launcher
// ---------------------------------------------------------------------------
extern "C" void kernel_launch(void* const* d_in, const int* in_sizes, int n_in,
                              void* d_out, int out_size, void* d_ws, size_t ws_size,
                              hipStream_t stream)
{
    (void)in_sizes; (void)n_in; (void)out_size; (void)ws_size;

    const float* x      = (const float*)d_in[0];
    const float* enc_w1 = (const float*)d_in[1];
    const float* enc_b1 = (const float*)d_in[2];
    const float* enc_w2 = (const float*)d_in[3];
    const float* enc_b2 = (const float*)d_in[4];
    const float* enc_w3 = (const float*)d_in[5];
    const float* enc_b3 = (const float*)d_in[6];
    const float* cb     = (const float*)d_in[7];
    const float* dec_w1 = (const float*)d_in[8];
    const float* dec_b1 = (const float*)d_in[9];
    const float* dec_w2 = (const float*)d_in[10];
    const float* dec_b2 = (const float*)d_in[11];
    const float* dec_w3 = (const float*)d_in[12];
    const float* dec_b3 = (const float*)d_in[13];

    char* ws = (char*)d_ws;
    float*          c2   = (float*)(ws + OFF_C2);
    float*          vqp  = (float*)(ws + OFF_VQP);
    float*          recp = (float*)(ws + OFF_RECP);
    unsigned short* wtE2 = (unsigned short*)(ws + OFF_WE2);
    unsigned short* wtE3 = (unsigned short*)(ws + OFF_WE3);
    unsigned short* wtD1 = (unsigned short*)(ws + OFF_WD1);
    unsigned short* wtD2 = (unsigned short*)(ws + OFF_WD2);
    unsigned short* h1   = (unsigned short*)(ws + OFF_H1);
    unsigned short* h2   = (unsigned short*)(ws + OFF_H2);
    float*          ze   = (float*)(ws + OFF_ZE);
    unsigned short* zq   = (unsigned short*)(ws + OFF_ZQ);
    unsigned short* d1   = (unsigned short*)(ws + OFF_D1);
    unsigned short* d2   = (unsigned short*)(ws + OFF_D2);

    // ----- tap tables -----
    auto packTap = [](int dy, int dx, int wt) -> unsigned long long {
        return (unsigned long long)(((dy + 1) & 3) | (((dx + 1) & 3) << 2) |
                                    ((wt & 15) << 4));
    };
    unsigned long long e2t0 = 0, e2t1 = 0;
    for (int ky = 0; ky < 4; ++ky)
        for (int kx = 0; kx < 4; ++kx) {
            int t = ky * 4 + kx;
            unsigned long long b = packTap(ky - 1, kx - 1, t);
            if (t < 8) e2t0 |= b << (8 * t); else e2t1 |= b << (8 * (t - 8));
        }
    unsigned long long e3t0 = 0, e3t1 = 0;
    for (int ky = 0; ky < 3; ++ky)
        for (int kx = 0; kx < 3; ++kx) {
            int t = ky * 3 + kx;
            unsigned long long b = packTap(ky - 1, kx - 1, t);
            if (t < 8) e3t0 |= b << (8 * t); else e3t1 |= b << (8 * (t - 8));
        }
    // ConvTranspose (k4,s2,p1) parity taps: parity 0 -> (dy,ky) in {(0,1),(-1,3)};
    // parity 1 -> {(1,0),(0,2)}
    int dys[2][2] = {{0, -1}, {1, 0}};
    int kys[2][2] = {{1, 3}, {0, 2}};
    unsigned long long dpt[2][2];
    for (int py = 0; py < 2; ++py)
        for (int px = 0; px < 2; ++px) {
            unsigned long long v = 0; int t = 0;
            for (int a = 0; a < 2; ++a)
                for (int c = 0; c < 2; ++c) {
                    v |= packTap(dys[py][a], dys[px][c],
                                 kys[py][a] * 4 + kys[px][c]) << (8 * t);
                    ++t;
                }
            dpt[py][px] = v;
        }

    // ----- pipeline -----
    wt_transform_kernel<<<1024, 256, 0, stream>>>(enc_w2, wtE2, 128, 128, 4, 0);
    wt_transform_kernel<<<288,  256, 0, stream>>>(enc_w3, wtE3, 128, 64,  3, 0);
    wt_transform_kernel<<<512,  256, 0, stream>>>(dec_w1, wtD1, 64,  128, 4, 1);
    wt_transform_kernel<<<1024, 256, 0, stream>>>(dec_w2, wtD2, 128, 128, 4, 1);
    cbnorm_kernel<<<2, 256, 0, stream>>>(cb, c2);

    // enc1: x -> h1 [64,64,64,128]
    enc1_kernel<<<131072, 256, 0, stream>>>(x, enc_w1, enc_b1, h1);

    // enc2: h1 -> h2 [64,32,32,128], k4 s2 p1, relu
    conv_wmma_kernel<<<512, 256, 0, stream>>>(
        h1, wtE2, enc_b2, h2, nullptr,
        64, 64, 128, /*lxBits*/5, /*bpiBits*/3,
        32, 32, 128,
        16, e2t0, e2t1, /*stride*/2, /*ostride*/1, 0, 0, /*relu*/1);

    // enc3: h2 -> z_e f32 [64,32,32,64], k3 s1 p1
    conv_wmma_kernel<<<512, 256, 0, stream>>>(
        h2, wtE3, enc_b3, nullptr, ze,
        32, 32, 128, 5, 3,
        32, 32, 64,
        9, e3t0, e3t1, 1, 1, 0, 0, 0);

    // VQ: z_e -> z_q bf16 + vq partials
    vq_kernel<<<1024, 128, 0, stream>>>(ze, cb, c2, zq, vqp);

    // dec1 (ConvTranspose 64->128): z_q -> d1 [64,64,64,128], 4 parity passes
    for (int py = 0; py < 2; ++py)
        for (int px = 0; px < 2; ++px)
            conv_wmma_kernel<<<512, 256, 0, stream>>>(
                zq, wtD1, dec_b1, d1, nullptr,
                32, 32, 64, 5, 3,
                64, 64, 128,
                4, dpt[py][px], 0ull, 1, 2, py, px, 1);

    // dec2 (ConvTranspose 128->128): d1 -> d2 [64,128,128,128]
    for (int py = 0; py < 2; ++py)
        for (int px = 0; px < 2; ++px)
            conv_wmma_kernel<<<2048, 256, 0, stream>>>(
                d1, wtD2, dec_b2, d2, nullptr,
                64, 64, 128, 6, 5,
                128, 128, 128,
                4, dpt[py][px], 0ull, 1, 2, py, px, 1);

    // dec3: d2 -> x_recon (d_out NCHW f32) + recon partials
    dec3_kernel<<<12288, 256, 0, stream>>>(d2, dec_w3, dec_b3, x,
                                           (float*)d_out, recp);

    // losses -> d_out tail
    finalize_kernel<<<1, 256, 0, stream>>>(vqp, recp, (float*)d_out + 3145728);
}
